// GIN_47459388621631
// MI455X (gfx1250) — matrix operand order repaired and checked by reading, not compiled, not run on hardware
//
#include <hip/hip_runtime.h>
#include <hip/hip_bf16.h>

typedef __attribute__((ext_vector_type(16))) _Float16 v16h;
typedef __attribute__((ext_vector_type(8)))  float    v8f;

#define N_NODES  50000
#define N_EDGES  800000
#define HID      128
#define N_LAYERS 4
#define N_GRAPHS 500
#define N_TARGETS 10

#define M_BLK        64                                  // rows per GEMM block
#define GEMM_BLOCKS  ((N_NODES + M_BLK - 1) / M_BLK)     // 782
#define PAD_ROWS     (GEMM_BLOCKS * M_BLK)               // 50048 (48 pad rows)

// ---------------------------------------------------------------------------
// agg[n][:] = cur[n][:]   (folds the GIN "+ h_i" (eps=0) self term)
// ---------------------------------------------------------------------------
__global__ void gin_copy_f4(const float* __restrict__ src, float* __restrict__ dst, int n4) {
    int i = blockIdx.x * blockDim.x + threadIdx.x;
    if (i < n4) ((float4*)dst)[i] = ((const float4*)src)[i];
}

__global__ void gin_zero(float* __restrict__ p, int n) {
    int i = blockIdx.x * blockDim.x + threadIdx.x;
    if (i < n) p[i] = 0.0f;
}

// ---------------------------------------------------------------------------
// Edge scatter: agg[dst[e]] += h[src[e]].  One wave32 per edge, float4/lane.
// Atomics are fire-and-forget (no return) -> STOREcnt, resolve in L2
// (77 MB working set < 192 MB L2).  Prefetch the gather row 16 edges ahead.
// ---------------------------------------------------------------------------
__global__ void gin_scatter(const float* __restrict__ h,
                            const int* __restrict__ esrc,
                            const int* __restrict__ edst,
                            float* __restrict__ agg) {
    int gid  = blockIdx.x * blockDim.x + threadIdx.x;
    int e    = gid >> 5;
    int lane = gid & 31;
    if (e >= N_EDGES) return;

    if (e + 16 < N_EDGES) {
        int sp = esrc[e + 16];
        __builtin_prefetch(h + (long)sp * HID + lane * 4, 0, 1);  // global_prefetch_b8
    }

    int s = esrc[e];
    int d = edst[e];
    float4 v = ((const float4*)(h + (long)s * HID))[lane];
    float* ap = agg + (long)d * HID + lane * 4;
    __hip_atomic_fetch_add(ap + 0, v.x, __ATOMIC_RELAXED, __HIP_MEMORY_SCOPE_AGENT);
    __hip_atomic_fetch_add(ap + 1, v.y, __ATOMIC_RELAXED, __HIP_MEMORY_SCOPE_AGENT);
    __hip_atomic_fetch_add(ap + 2, v.z, __ATOMIC_RELAXED, __HIP_MEMORY_SCOPE_AGENT);
    __hip_atomic_fetch_add(ap + 3, v.w, __ATOMIC_RELAXED, __HIP_MEMORY_SCOPE_AGENT);
}

// ---------------------------------------------------------------------------
// Weight pre-pack: W [128x128] f32 (row-major K x N) -> f16 fragments in the
// per-lane WMMA B layout in contiguous-K order (k = 16*(lane>>4) + j): fragment entry
//   i = (kc*8 + colTile)*32 + lane ;  element j = W[kb+j][colTile*16 + n]
//   with n = lane&15, kb = kc*32 + (lane>>4)*16.
// One block per matrix (8 matrices: 4 layers x {W1,W2}); 256 threads, 4
// fragment entries each.  32 B contiguous store per entry.
// ---------------------------------------------------------------------------
__global__ void gin_pack_w(const float* __restrict__ conv_w1,
                           const float* __restrict__ conv_w2,
                           _Float16* __restrict__ pack) {
    int m = blockIdx.x;                                    // 0..7
    const float* W = (m < 4) ? (conv_w1 + (long)m * HID * HID)
                             : (conv_w2 + (long)(m - 4) * HID * HID);
    _Float16* P = pack + (long)m * (HID * HID);            // 16384 halves / matrix
#pragma unroll
    for (int rep = 0; rep < 4; ++rep) {
        int i    = threadIdx.x + rep * 256;                // 0..1023
        int kc   = i >> 8;
        int ct   = (i >> 5) & 7;
        int lane = i & 31;
        int n  = lane & 15;
        int kb = kc * 32 + (lane >> 4) * 16;
        const float* s = W + (long)kb * HID + ct * 16 + n;
        v16h f;
#pragma unroll
        for (int j = 0; j < 16; ++j) f[j] = (_Float16)s[(long)j * HID];
        *(v16h*)(P + (long)i * 16) = f;
    }
}

// ---------------------------------------------------------------------------
// Y = act(X @ W + bias), X:[PAD_ROWS x 128] f32 (ws buffer), Wp: packed f16.
// grid = 782 blocks x 64 rows, block = 256 (8 wave32).
//  - 256 threads stage the 64x128 A tile into 16 KB LDS as f16 in fragment
//    order (A layout: lane<16 -> row=lane, K={0..15};
//    lane>=16 -> row=lane-16, K={16..31}).
//  - Wave w owns columns [16w,16w+16): loads its 4 B fragments (32 VGPRs)
//    once from the packed weights, then sweeps 4 M-tiles: per tile a chain
//    of 4 v_wmma_f32_16x16x32_f16 fed by ds_load_b128 (16 WMMA / wave).
// ---------------------------------------------------------------------------
__global__ void gin_gemm_wmma(const float* __restrict__ X,
                              const _Float16* __restrict__ Wp,
                              const float* __restrict__ bias,
                              float* __restrict__ Y,
                              int relu) {
    __shared__ __align__(32) _Float16 As[4 * 4 * 32 * 16];   // 16 KB [mt][kc][lane][16]

    int tid  = threadIdx.x;
    int wave = tid >> 5;              // 0..7 == column tile
    int lane = tid & 31;
    long rowBase = (long)blockIdx.x * M_BLK;

    // --- cooperative A-tile stage: 512 fragment entries, 2 per thread ---
#pragma unroll
    for (int rep = 0; rep < 2; ++rep) {
        int i  = tid + rep * 256;                 // (mt*4 + kc)*32 + ln
        int mt = i >> 7;
        int kc = (i >> 5) & 3;
        int ln = i & 31;
        int hs = ln >> 4;
        const float* p = X + (rowBase + mt * 16 + (ln & 15)) * HID + kc * 32;
        _Float16* d = &As[(long)i * 16];
#pragma unroll
        for (int t = 0; t < 8; ++t) {
            d[t]     = (_Float16)p[hs * 16 + t];
            d[t + 8] = (_Float16)p[hs * 16 + 8 + t];
        }
    }
    __syncthreads();

    // --- B fragments for this wave's column tile: load once, keep in VGPRs ---
    v16h b0 = *(const v16h*)(Wp + (long)((0 * 8 + wave) * 32 + lane) * 16);
    v16h b1 = *(const v16h*)(Wp + (long)((1 * 8 + wave) * 32 + lane) * 16);
    v16h b2 = *(const v16h*)(Wp + (long)((2 * 8 + wave) * 32 + lane) * 16);
    v16h b3 = *(const v16h*)(Wp + (long)((3 * 8 + wave) * 32 + lane) * 16);

    int col  = wave * 16 + (lane & 15);
    float bv = bias[col];
    int rOff = (lane >> 4) ? 8 : 0;   // C layout: VGPR r -> M = r + 8*(lane>>4)

#pragma unroll
    for (int mt = 0; mt < 4; ++mt) {
        const _Float16* ab = &As[(long)(mt * 4) * 32 * 16 + (long)lane * 16];
        v8f c = {};
        c = __builtin_amdgcn_wmma_f32_16x16x32_f16(false, *(const v16h*)(ab + 0 * 512),
                                                   false, b0, (short)0, c, false, false);
        c = __builtin_amdgcn_wmma_f32_16x16x32_f16(false, *(const v16h*)(ab + 1 * 512),
                                                   false, b1, (short)0, c, false, false);
        c = __builtin_amdgcn_wmma_f32_16x16x32_f16(false, *(const v16h*)(ab + 2 * 512),
                                                   false, b2, (short)0, c, false, false);
        c = __builtin_amdgcn_wmma_f32_16x16x32_f16(false, *(const v16h*)(ab + 3 * 512),
                                                   false, b3, (short)0, c, false, false);

        long row0 = rowBase + mt * 16 + rOff;
#pragma unroll
        for (int r = 0; r < 8; ++r) {
            float v = c[r] + bv;
            if (relu) v = fmaxf(v, 0.0f);
            Y[(row0 + r) * HID + col] = v;
        }
    }
}

// ---------------------------------------------------------------------------
// global_add_pool: g[batch[n]] += h[n].  One wave per node, float4/lane.
// ---------------------------------------------------------------------------
__global__ void gin_pool(const float* __restrict__ h,
                         const int* __restrict__ batch,
                         float* __restrict__ g) {
    int gid  = blockIdx.x * blockDim.x + threadIdx.x;
    int n    = gid >> 5;
    int lane = gid & 31;
    if (n >= N_NODES) return;
    int b = batch[n];
    float4 v = ((const float4*)(h + (long)n * HID))[lane];
    float* gp = g + (long)b * HID + lane * 4;
    __hip_atomic_fetch_add(gp + 0, v.x, __ATOMIC_RELAXED, __HIP_MEMORY_SCOPE_AGENT);
    __hip_atomic_fetch_add(gp + 1, v.y, __ATOMIC_RELAXED, __HIP_MEMORY_SCOPE_AGENT);
    __hip_atomic_fetch_add(gp + 2, v.z, __ATOMIC_RELAXED, __HIP_MEMORY_SCOPE_AGENT);
    __hip_atomic_fetch_add(gp + 3, v.w, __ATOMIC_RELAXED, __HIP_MEMORY_SCOPE_AGENT);
}

// ---------------------------------------------------------------------------
// Readout head: out = relu(g @ W1 + b1) @ W2 + b2.  One block per graph.
// ---------------------------------------------------------------------------
__global__ void gin_head(const float* __restrict__ g,
                         const float* __restrict__ w1, const float* __restrict__ b1,
                         const float* __restrict__ w2, const float* __restrict__ b2,
                         float* __restrict__ out) {
    __shared__ float gs[HID];
    __shared__ float ts[64];
    int gi = blockIdx.x;
    int t  = threadIdx.x;             // 128 threads
    gs[t] = g[(long)gi * HID + t];
    __syncthreads();
    if (t < 64) {
        float acc = b1[t];
#pragma unroll 8
        for (int k = 0; k < HID; ++k) acc += gs[k] * w1[(long)k * 64 + t];
        ts[t] = fmaxf(acc, 0.0f);
    }
    __syncthreads();
    if (t < N_TARGETS) {
        float acc = b2[t];
#pragma unroll
        for (int k = 0; k < 64; ++k) acc += ts[k] * w2[(long)k * N_TARGETS + t];
        out[(long)gi * N_TARGETS + t] = acc;
    }
}

// ---------------------------------------------------------------------------
extern "C" void kernel_launch(void* const* d_in, const int* in_sizes, int n_in,
                              void* d_out, int out_size, void* d_ws, size_t ws_size,
                              hipStream_t stream) {
    const float* x       = (const float*)d_in[0];
    const int*   eidx    = (const int*)d_in[1];      // [2, N_EDGES]
    const int*   batch   = (const int*)d_in[2];      // [N_NODES]
    const float* conv_w1 = (const float*)d_in[3];    // [4,128,128]
    const float* conv_b1 = (const float*)d_in[4];    // [4,128]
    const float* conv_w2 = (const float*)d_in[5];    // [4,128,128]
    const float* conv_b2 = (const float*)d_in[6];    // [4,128]
    const float* head_w1 = (const float*)d_in[7];    // [128,64]
    const float* head_b1 = (const float*)d_in[8];    // [64]
    const float* head_w2 = (const float*)d_in[9];    // [64,10]
    const float* head_b2 = (const float*)d_in[10];   // [10]
    float* out = (float*)d_out;

    const int* esrc = eidx;
    const int* edst = eidx + N_EDGES;

    // Feature buffers padded to 50048 rows so every GEMM block has 64 full
    // rows (pad rows hold deterministic junk; nothing downstream reads them).
    const size_t padBytes = (size_t)PAD_ROWS * HID * sizeof(float);   // ~25.6 MB
    char* ws   = (char*)d_ws;
    float*     agg   = (float*)(ws);
    float*     z     = (float*)(ws + padBytes);
    float*     hb    = (float*)(ws + 2 * padBytes);
    float*     g     = (float*)(ws + 3 * padBytes);                   // [500,128]
    _Float16*  wpack = (_Float16*)(ws + 3 * padBytes
                                      + (size_t)N_GRAPHS * HID * sizeof(float));
    // wpack: 8 matrices x 16384 halves = 256 KB

    const int nFeat4 = N_NODES * HID / 4;                              // 1.6M float4
    dim3 blk(256);
    dim3 gridCopy((nFeat4 + 255) / 256);
    dim3 gridScat((N_EDGES * 32) / 256);                               // wave per edge
    dim3 gridGemm(GEMM_BLOCKS);
    dim3 gridPool((N_NODES * 32 + 255) / 256);

    // Pack all conv weights into f16 WMMA-B fragment order (every call:
    // deterministic, graph-capture safe, 256 KB total -> L2 resident).
    gin_pack_w<<<8, blk, 0, stream>>>(conv_w1, conv_w2, wpack);

    const float* cur = x;
    for (int layer = 0; layer < N_LAYERS; ++layer) {
        // agg = cur  (self term), then agg += sum over incoming edges
        gin_copy_f4<<<gridCopy, blk, 0, stream>>>(cur, agg, nFeat4);
        gin_scatter<<<gridScat, blk, 0, stream>>>(cur, esrc, edst, agg);
        // z = relu(agg @ W1 + b1); h = z @ W2 + b2
        gin_gemm_wmma<<<gridGemm, blk, 0, stream>>>(
            agg, wpack + (long)layer * HID * HID, conv_b1 + layer * HID, z, 1);
        gin_gemm_wmma<<<gridGemm, blk, 0, stream>>>(
            z, wpack + (long)(4 + layer) * HID * HID, conv_b2 + layer * HID, hb, 0);
        cur = hb;
    }

    gin_zero<<<(N_GRAPHS * HID + 255) / 256, blk, 0, stream>>>(g, N_GRAPHS * HID);
    gin_pool<<<gridPool, blk, 0, stream>>>(cur, batch, g);
    gin_head<<<N_GRAPHS, 128, 0, stream>>>(g, head_w1, head_b1, head_w2, head_b2, out);
}
